// SwinTransformerBlock_89816356094360
// MI455X (gfx1250) — matrix operand order repaired and checked by reading, not compiled, not run on hardware
//
#include <hip/hip_runtime.h>
#include <hip/hip_bf16.h>
#include <math.h>

#define HH      56
#define WSZ     7
#define SHIFT_  3
#define HEADS_  6
#define DIM_    192
#define HID_    768
#define NTOK    49          // tokens per window
#define TOTWIN  2048        // 32 batches * 64 windows
#define MTOK    100352      // TOTWIN * NTOK
#define QSCALE  0.17677669529663687f   // 1/sqrt(32)

typedef __bf16 bf16;
typedef __attribute__((ext_vector_type(16))) bf16  v16bf;
typedef __attribute__((ext_vector_type(8)))  bf16  v8bf;
typedef __attribute__((ext_vector_type(8)))  float v8f;

// ---------------------------------------------------------------------------
// WMMA fragment loaders (CDNA5 ISA 7.12.2, wave32)
// A 16x32 bf16: lane = m(0..15)|c<<4 ; elems 0..7 -> k=c*8.., 8..15 -> k=c*8+16..
// B 32x16 bf16: lane = n(0..15)|c<<4 ; elems 0..7 -> k=c*8.., 8..15 -> k=c*8+16..
// C/D 16x16 f32: row m = 8*(lane>>4)+v, col n = lane&15
// ---------------------------------------------------------------------------
static __device__ __forceinline__ v16bf load_frag_a(const bf16* base, int ld) {
  const int lane = threadIdx.x & 31;
  const int m = lane & 15, c = lane >> 4;
  const bf16* p = base + (size_t)m * ld + c * 8;
  union { v16bf v; v8bf h[2]; } u;
  u.h[0] = *reinterpret_cast<const v8bf*>(p);
  u.h[1] = *reinterpret_cast<const v8bf*>(p + 16);
  return u.v;
}

static __device__ __forceinline__ v16bf load_frag_b(const bf16* base, int ld) {
  const int lane = threadIdx.x & 31;
  const int n = lane & 15, c = lane >> 4;
  const bf16* p = base + (size_t)(c * 8) * ld + n;
  v16bf v;
  #pragma unroll
  for (int s = 0; s < 8; s++) {
    v[s] = p[(size_t)s * ld];
    v[8 + s] = p[(size_t)(16 + s) * ld];
  }
  return v;
}

// 32x32 output macro-tile: 2x2 WMMA tiles, A from LDS, B (weights) from global/L2
static __device__ __forceinline__ void gemm32x32(const bf16* Alds, int lda,
                                                 const bf16* Bg, int ldb, int K,
                                                 v8f acc[2][2]) {
  for (int k0 = 0; k0 < K; k0 += 32) {
    if (k0 + 32 < K) __builtin_prefetch(Bg + (size_t)(k0 + 32) * ldb, 0, 0);
    v16bf a0 = load_frag_a(Alds + k0, lda);
    v16bf a1 = load_frag_a(Alds + (size_t)16 * lda + k0, lda);
    v16bf b0 = load_frag_b(Bg + (size_t)k0 * ldb, ldb);
    v16bf b1 = load_frag_b(Bg + (size_t)k0 * ldb + 16, ldb);
    acc[0][0] = __builtin_amdgcn_wmma_f32_16x16x32_bf16(false, a0, false, b0, (short)0, acc[0][0], false, false);
    acc[0][1] = __builtin_amdgcn_wmma_f32_16x16x32_bf16(false, a0, false, b1, (short)0, acc[0][1], false, false);
    acc[1][0] = __builtin_amdgcn_wmma_f32_16x16x32_bf16(false, a1, false, b0, (short)0, acc[1][0], false, false);
    acc[1][1] = __builtin_amdgcn_wmma_f32_16x16x32_bf16(false, a1, false, b1, (short)0, acc[1][1], false, false);
  }
}

// ---------------------------------------------------------------------------
// fp32 -> bf16 weight conversion
// ---------------------------------------------------------------------------
__global__ __launch_bounds__(256) void cvt_kernel(const float* __restrict__ s,
                                                  bf16* __restrict__ d, int n) {
  int i = blockIdx.x * 256 + threadIdx.x;
  if (i < n) d[i] = (bf16)s[i];
}

// ---------------------------------------------------------------------------
// Precompute additive softmax table: T[sel][h][m][n], sel = (wy==7)*2|(wx==7).
// Contains rel-pos bias + shifted-window mask; -1e9 for padded rows/cols.
// ---------------------------------------------------------------------------
__global__ __launch_bounds__(256) void bias_table_kernel(const float* __restrict__ rpb,
                                                         float* __restrict__ T) {
  int i = blockIdx.x * 256 + threadIdx.x;
  if (i >= 4 * HEADS_ * 64 * 64) return;
  int n = i & 63, m = (i >> 6) & 63;
  int h = (i >> 12) % HEADS_, sel = (i >> 12) / HEADS_;
  float v = -1e9f;
  if (m < NTOK && n < NTOK) {
    int tym = m / WSZ, txm = m % WSZ, tyn = n / WSZ, txn = n % WSZ;
    int ridx = (tym - tyn + WSZ - 1) * (2 * WSZ - 1) + (txm - txn + WSZ - 1);
    v = rpb[ridx * HEADS_ + h];
    int wy7 = sel >> 1, wx7 = sel & 1;   // interior windows (wy<7) are all region 0
    int rym = wy7 ? (tym < 4 ? 1 : 2) : 0;
    int ryn = wy7 ? (tyn < 4 ? 1 : 2) : 0;
    int rxm = wx7 ? (txm < 4 ? 1 : 2) : 0;
    int rxn = wx7 ? (txn < 4 ? 1 : 2) : 0;
    if (rym != ryn || rxm != rxn) v -= 100.f;
  }
  T[i] = v;
}

// ---------------------------------------------------------------------------
// Fused per-window kernel: LN1 + shift/partition + QKV proj + attention +
// out proj + window reverse + residual.  One block (8 waves) per window.
// All intermediates live in LDS (112 KB of the WGP's 320 KB).
// ---------------------------------------------------------------------------
__global__ __launch_bounds__(256) void attn_mega_kernel(
    const float* __restrict__ x, const float* __restrict__ n1g, const float* __restrict__ n1b,
    const bf16* __restrict__ Wqkv, const float* __restrict__ qkvb,
    const bf16* __restrict__ Wproj, const float* __restrict__ projb,
    const float* __restrict__ T, float* __restrict__ x1) {
  __shared__ __align__(16) bf16 smem[57344];          // 112 KB
  bf16* Xin = smem;                                    // [64][192] LN'd inputs
  bf16* Os  = smem;                                    // alias: attention output (Xin dead)
  bf16* Qs  = smem + 12288;                            // [64][192]
  bf16* Kt  = smem + 24576;                            // [192][64] (K transposed)
  bf16* Vs  = smem + 36864;                            // [64][192]
  bf16* Ps  = smem + 49152;                            // [8 waves][16][64] softmax scratch

  const int w = blockIdx.x;
  const int b = w >> 6, wy = (w >> 3) & 7, wx = w & 7;
  const int tid = threadIdx.x, wv = tid >> 5, lane = tid & 31;
  const int nloc = lane & 15, mhalf = (lane >> 4) * 8;

  // ---- stage A: LN1 + cyclic shift + window gather (one wave per token) ----
  for (int t = wv; t < 64; t += 8) {
    if (t < NTOK) {
      int y = wy * WSZ + t / WSZ + SHIFT_; if (y >= HH) y -= HH;
      int xx = wx * WSZ + t % WSZ + SHIFT_; if (xx >= HH) xx -= HH;
      size_t src = ((size_t)b * (HH * HH) + (size_t)y * HH + xx) * DIM_;
      float vals[6]; float s = 0.f;
      #pragma unroll
      for (int i = 0; i < 6; i++) { vals[i] = x[src + i * 32 + lane]; s += vals[i]; }
      #pragma unroll
      for (int d = 16; d >= 1; d >>= 1) s += __shfl_xor(s, d, 32);
      float mean = s * (1.0f / DIM_);
      float vsum = 0.f;
      #pragma unroll
      for (int i = 0; i < 6; i++) { float d = vals[i] - mean; vsum += d * d; }
      #pragma unroll
      for (int d = 16; d >= 1; d >>= 1) vsum += __shfl_xor(vsum, d, 32);
      float rstd = rsqrtf(vsum * (1.0f / DIM_) + 1e-5f);
      #pragma unroll
      for (int i = 0; i < 6; i++) {
        int c = i * 32 + lane;
        Xin[t * DIM_ + c] = (bf16)((vals[i] - mean) * rstd * n1g[c] + n1b[c]);
      }
    } else {
      #pragma unroll
      for (int i = 0; i < 6; i++) Xin[t * DIM_ + i * 32 + lane] = (bf16)0.f;
    }
  }
  __syncthreads();

  // ---- stage B: QKV = Xin @ Wqkv (+bias, q scaled), scattered into LDS ----
  for (int g = wv; g < 36; g += 8) {                  // 2 (M) x 18 (N) macro-tiles
    int r0 = (g & 1) * 32, c0 = (g >> 1) * 32;
    v8f acc[2][2] = {};
    gemm32x32(Xin + r0 * DIM_, DIM_, Wqkv + c0, 3 * DIM_, DIM_, acc);
    #pragma unroll
    for (int i = 0; i < 2; i++)
      #pragma unroll
      for (int j = 0; j < 2; j++)
        #pragma unroll
        for (int v = 0; v < 8; v++) {
          int m = r0 + i * 16 + mhalf + v;
          int n = c0 + j * 16 + nloc;
          float val = acc[i][j][v] + qkvb[n];
          if (n < DIM_)          Qs[m * DIM_ + n] = (bf16)(val * QSCALE);
          else if (n < 2 * DIM_) Kt[(n - DIM_) * 64 + m] = (bf16)val;
          else                   Vs[m * DIM_ + (n - 2 * DIM_)] = (bf16)val;
        }
  }
  __syncthreads();

  // ---- stage C: attention (jobs = (head, row-block), 3 per wave) ----
  const float* Tsel = T + (size_t)(((wy == 7) ? 2 : 0) | ((wx == 7) ? 1 : 0)) * (HEADS_ * 4096);
  bf16* Pw = Ps + wv * 1024;
  for (int job = wv; job < 24; job += 8) {
    int h = job % HEADS_, rb = job / HEADS_;
    const float* Th = Tsel + h * 4096;
    v16bf qf = load_frag_a(Qs + rb * 16 * DIM_ + h * 32, DIM_);
    v8f S[4];
    #pragma unroll
    for (int cb = 0; cb < 4; cb++) {
      v16bf kf = load_frag_b(Kt + h * 32 * 64 + cb * 16, 64);
      v8f z = {};
      S[cb] = __builtin_amdgcn_wmma_f32_16x16x32_bf16(false, qf, false, kf, (short)0, z, false, false);
    }
    #pragma unroll
    for (int v = 0; v < 8; v++) {                    // branch-free bias+mask, softmax
      const float* Trow = Th + (rb * 16 + mhalf + v) * 64 + nloc;
      #pragma unroll
      for (int cb = 0; cb < 4; cb++) S[cb][v] += Trow[cb * 16];
      float mx = fmaxf(fmaxf(S[0][v], S[1][v]), fmaxf(S[2][v], S[3][v]));
      #pragma unroll
      for (int d = 8; d >= 1; d >>= 1) mx = fmaxf(mx, __shfl_xor(mx, d, 16));
      float sum = 0.f;
      #pragma unroll
      for (int cb = 0; cb < 4; cb++) { float e = __expf(S[cb][v] - mx); S[cb][v] = e; sum += e; }
      #pragma unroll
      for (int d = 8; d >= 1; d >>= 1) sum += __shfl_xor(sum, d, 16);
      float inv = 1.0f / sum;
      #pragma unroll
      for (int cb = 0; cb < 4; cb++)
        Pw[(mhalf + v) * 64 + cb * 16 + nloc] = (bf16)(S[cb][v] * inv);  // C->A relayout via LDS
    }
    #pragma unroll
    for (int nb = 0; nb < 2; nb++) {                 // O = P @ V (padded P cols are exact 0)
      v8f o = {};
      #pragma unroll
      for (int kk = 0; kk < 2; kk++) {
        v16bf pf = load_frag_a(Pw + kk * 32, 64);
        v16bf vf = load_frag_b(Vs + kk * 32 * DIM_ + h * 32 + nb * 16, DIM_);
        o = __builtin_amdgcn_wmma_f32_16x16x32_bf16(false, pf, false, vf, (short)0, o, false, false);
      }
      #pragma unroll
      for (int v = 0; v < 8; v++)
        Os[(rb * 16 + mhalf + v) * DIM_ + h * 32 + nb * 16 + nloc] = (bf16)o[v];
    }
  }
  __syncthreads();

  // ---- stage D: out proj + window reverse + reverse shift + residual ----
  for (int g = wv; g < 12; g += 8) {                  // 2 (M) x 6 (N) macro-tiles
    int r0 = (g & 1) * 32, c0 = (g >> 1) * 32;
    v8f acc[2][2] = {};
    gemm32x32(Os + r0 * DIM_, DIM_, Wproj + c0, DIM_, DIM_, acc);
    #pragma unroll
    for (int i = 0; i < 2; i++)
      #pragma unroll
      for (int j = 0; j < 2; j++)
        #pragma unroll
        for (int v = 0; v < 8; v++) {
          int m = r0 + i * 16 + mhalf + v;
          if (m < NTOK) {
            int n = c0 + j * 16 + nloc;
            int y = wy * WSZ + m / WSZ + SHIFT_; if (y >= HH) y -= HH;
            int xx = wx * WSZ + m % WSZ + SHIFT_; if (xx >= HH) xx -= HH;
            size_t idx = ((size_t)b * (HH * HH) + (size_t)y * HH + xx) * DIM_ + n;
            x1[idx] = x[idx] + acc[i][j][v] + projb[n];
          }
        }
  }
}

// ---------------------------------------------------------------------------
// Fused MLP kernel: LN2 + fc1 + exact GELU + fc2 + residual for a 64-token
// strip.  Hidden activations (64x768 bf16 = 96 KB) never leave LDS.
// ---------------------------------------------------------------------------
__global__ __launch_bounds__(256) void mlp_mega_kernel(
    const float* __restrict__ x1, const float* __restrict__ n2g, const float* __restrict__ n2b,
    const bf16* __restrict__ W1, const float* __restrict__ b1,
    const bf16* __restrict__ W2, const float* __restrict__ b2,
    float* __restrict__ out) {
  __shared__ __align__(16) bf16 smem[61440];          // 120 KB
  bf16* XA = smem;                                    // [64][192]
  bf16* Hs = smem + 12288;                            // [64][768]
  const int row0 = blockIdx.x * 64;
  const int tid = threadIdx.x, wv = tid >> 5, lane = tid & 31;
  const int nloc = lane & 15, mhalf = (lane >> 4) * 8;

  // ---- stage A: LN2 (one wave per token) ----
  for (int lt = wv; lt < 64; lt += 8) {
    size_t src = (size_t)(row0 + lt) * DIM_;
    float vals[6]; float s = 0.f;
    #pragma unroll
    for (int i = 0; i < 6; i++) { vals[i] = x1[src + i * 32 + lane]; s += vals[i]; }
    #pragma unroll
    for (int d = 16; d >= 1; d >>= 1) s += __shfl_xor(s, d, 32);
    float mean = s * (1.0f / DIM_);
    float vsum = 0.f;
    #pragma unroll
    for (int i = 0; i < 6; i++) { float d = vals[i] - mean; vsum += d * d; }
    #pragma unroll
    for (int d = 16; d >= 1; d >>= 1) vsum += __shfl_xor(vsum, d, 32);
    float rstd = rsqrtf(vsum * (1.0f / DIM_) + 1e-5f);
    #pragma unroll
    for (int i = 0; i < 6; i++) {
      int c = i * 32 + lane;
      XA[lt * DIM_ + c] = (bf16)((vals[i] - mean) * rstd * n2g[c] + n2b[c]);
    }
  }
  __syncthreads();

  // ---- stage B: H = gelu(XA @ W1 + b1) ----
  for (int g = wv; g < 48; g += 8) {                  // 2 (M) x 24 (N) macro-tiles
    int r0 = (g & 1) * 32, c0 = (g >> 1) * 32;
    v8f acc[2][2] = {};
    gemm32x32(XA + r0 * DIM_, DIM_, W1 + c0, HID_, DIM_, acc);
    #pragma unroll
    for (int i = 0; i < 2; i++)
      #pragma unroll
      for (int j = 0; j < 2; j++)
        #pragma unroll
        for (int v = 0; v < 8; v++) {
          int m = r0 + i * 16 + mhalf + v;
          int n = c0 + j * 16 + nloc;
          float val = acc[i][j][v] + b1[n];
          val = 0.5f * val * (1.0f + erff(val * 0.70710678118654752f));
          Hs[m * HID_ + n] = (bf16)val;
        }
  }
  __syncthreads();

  // ---- stage C: out = Hs @ W2 + b2 + x1 ----
  for (int g = wv; g < 12; g += 8) {                  // 2 (M) x 6 (N) macro-tiles
    int r0 = (g & 1) * 32, c0 = (g >> 1) * 32;
    v8f acc[2][2] = {};
    gemm32x32(Hs + r0 * HID_, HID_, W2 + c0, DIM_, HID_, acc);
    #pragma unroll
    for (int i = 0; i < 2; i++)
      #pragma unroll
      for (int j = 0; j < 2; j++)
        #pragma unroll
        for (int v = 0; v < 8; v++) {
          size_t idx = (size_t)(row0 + r0 + i * 16 + mhalf + v) * DIM_ + c0 + j * 16 + nloc;
          out[idx] = x1[idx] + acc[i][j][v] + b2[c0 + j * 16 + nloc];
        }
  }
}

// ---------------------------------------------------------------------------
extern "C" void kernel_launch(void* const* d_in, const int* in_sizes, int n_in,
                              void* d_out, int out_size, void* d_ws, size_t ws_size,
                              hipStream_t stream) {
  (void)in_sizes; (void)n_in; (void)out_size; (void)ws_size;
  const float* x     = (const float*)d_in[0];
  const float* n1g   = (const float*)d_in[1];
  const float* n1b   = (const float*)d_in[2];
  const float* qkvw  = (const float*)d_in[3];
  const float* qkvb  = (const float*)d_in[4];
  const float* rpb   = (const float*)d_in[5];
  const float* projw = (const float*)d_in[6];
  const float* projb = (const float*)d_in[7];
  const float* n2g   = (const float*)d_in[8];
  const float* n2b   = (const float*)d_in[9];
  const float* w1    = (const float*)d_in[10];
  const float* b1    = (const float*)d_in[11];
  const float* w2    = (const float*)d_in[12];
  const float* b2    = (const float*)d_in[13];
  float* out = (float*)d_out;

  char* ws = (char*)d_ws;
  size_t off = 0;
  auto take = [&](size_t bytes) -> char* {
    char* p = ws + off;
    off += (bytes + 255) & ~(size_t)255;
    return p;
  };
  bf16*  wqkv  = (bf16*)take((size_t)DIM_ * 3 * DIM_ * sizeof(bf16));
  bf16*  wproj = (bf16*)take((size_t)DIM_ * DIM_ * sizeof(bf16));
  bf16*  wm1   = (bf16*)take((size_t)DIM_ * HID_ * sizeof(bf16));
  bf16*  wm2   = (bf16*)take((size_t)HID_ * DIM_ * sizeof(bf16));
  float* Tbl   = (float*)take((size_t)4 * HEADS_ * 64 * 64 * sizeof(float));
  float* x1    = (float*)take((size_t)MTOK * DIM_ * sizeof(float));

  cvt_kernel<<<(DIM_ * 3 * DIM_ + 255) / 256, 256, 0, stream>>>(qkvw, wqkv, DIM_ * 3 * DIM_);
  cvt_kernel<<<(DIM_ * DIM_ + 255) / 256, 256, 0, stream>>>(projw, wproj, DIM_ * DIM_);
  cvt_kernel<<<(DIM_ * HID_ + 255) / 256, 256, 0, stream>>>(w1, wm1, DIM_ * HID_);
  cvt_kernel<<<(HID_ * DIM_ + 255) / 256, 256, 0, stream>>>(w2, wm2, HID_ * DIM_);
  bias_table_kernel<<<(4 * HEADS_ * 64 * 64 + 255) / 256, 256, 0, stream>>>(rpb, Tbl);

  // fused attention half: x -> x1 (one block per window)
  attn_mega_kernel<<<TOTWIN, 256, 0, stream>>>(x, n1g, n1b, wqkv, qkvb,
                                               wproj, projb, Tbl, x1);
  // fused MLP half: x1 -> out (one block per 64-token strip)
  mlp_mega_kernel<<<MTOK / 64, 256, 0, stream>>>(x1, n2g, n2b, wm1, b1, wm2, b2, out);
}
